// CondConv2d_70222715289934
// MI455X (gfx1250) — matrix operand order repaired and checked by reading, not compiled, not run on hardware
//
#include <hip/hip_runtime.h>
#include <hip/hip_bf16.h>

// ---------- types ----------
typedef __attribute__((ext_vector_type(16))) __bf16 v16bf;
typedef __attribute__((ext_vector_type(8)))  float  v8f;
typedef __attribute__((ext_vector_type(8)))  unsigned int v8u;
typedef __attribute__((ext_vector_type(4)))  unsigned int v4u;
typedef __attribute__((ext_vector_type(8)))  int v8i;
typedef __attribute__((ext_vector_type(4)))  int v4i;

// ---------- problem constants ----------
#define BS    64
#define CIN   128
#define HW_IN 56
#define NPIX  (HW_IN * HW_IN)      // 3136
#define KEXP  8
#define COUT  128
#define KTOT  (CIN * 9)            // 1152
#define WMIX_PER_B (COUT * KTOT)   // 147456

#if defined(__has_builtin)
#if __has_builtin(__builtin_amdgcn_tensor_load_to_lds) && __has_builtin(__builtin_amdgcn_s_wait_tensorcnt)
#define HAVE_TDM 1
#endif
#endif

// fp32 -> bf16, round-to-nearest-even
__device__ __forceinline__ unsigned short f2bf(float f) {
    unsigned u = __builtin_bit_cast(unsigned, f);
    u = (u + 0x7FFFu + ((u >> 16) & 1u)) >> 16;
    return (unsigned short)u;
}

// ---------- kernel 1: per-sample expert-mixed weights (bf16) ----------
__global__ __launch_bounds__(256) void mix_weights_kernel(
    const float* __restrict__ routing,   // [BS, KEXP]
    const float* __restrict__ weight,    // [KEXP, COUT, CIN, 3, 3]
    unsigned short* __restrict__ wmix)   // [BS, COUT, KTOT] bf16
{
    const int b   = blockIdx.y;
    const int idx = blockIdx.x * 256 + threadIdx.x;   // < WMIX_PER_B
    float acc = 0.f;
#pragma unroll
    for (int e = 0; e < KEXP; ++e)
        acc += routing[b * KEXP + e] * weight[(size_t)e * WMIX_PER_B + idx];
    wmix[(size_t)b * WMIX_PER_B + idx] = f2bf(acc);
}

// ---------- kernel 2: x fp32 -> bf16 (4 elems/thread) ----------
__global__ __launch_bounds__(256) void x_to_bf16_kernel(
    const float* __restrict__ x, unsigned short* __restrict__ xbf)
{
    const size_t i4 = ((size_t)blockIdx.x * 256 + threadIdx.x) * 4;
    float4 v = *(const float4*)(x + i4);
    uint2 r;
    r.x = (unsigned)f2bf(v.x) | ((unsigned)f2bf(v.y) << 16);
    r.y = (unsigned)f2bf(v.z) | ((unsigned)f2bf(v.w) << 16);
    *(uint2*)(xbf + i4) = r;
}

// ---------- kernel 3: mixed bias (fp32) ----------
__global__ __launch_bounds__(256) void mix_bias_kernel(
    const float* __restrict__ routing,  // [BS, KEXP]
    const float* __restrict__ bias,     // [KEXP, COUT]
    float* __restrict__ bmix)           // [BS, COUT]
{
    const int i  = blockIdx.x * 256 + threadIdx.x;   // < BS*COUT
    const int b  = i >> 7;
    const int oc = i & 127;
    float acc = 0.f;
#pragma unroll
    for (int e = 0; e < KEXP; ++e)
        acc += routing[b * KEXP + e] * bias[e * COUT + oc];
    bmix[i] = acc;
}

// ---------- kernel 4: implicit-GEMM conv via bf16 WMMA + TDM ----------
// Block: full M=128 oc x 64 pixels of one sample. 8 waves as 4(M) x 2(N);
// each wave owns 32(M) x 32(N) -> 4 WMMAs per 32-wide K step.
// A tile staged by the Tensor Data Mover directly into padded LDS rows;
// B tile is a manual im2col gather. Double-buffered LDS, 1 barrier/step.
#define ASTRIDE 48                 // 32 k elems + 16 pad -> 96B rows (16B aligned)
#define ASZ (128 * ASTRIDE)
#define BSZ (64 * ASTRIDE)

__global__ __launch_bounds__(256) void condconv_wmma_kernel(
    const unsigned short* __restrict__ xbf,   // [BS, CIN, 56, 56] bf16
    const unsigned short* __restrict__ wmix,  // [BS, COUT, KTOT] bf16
    const float* __restrict__ bmix,           // [BS, COUT]
    float* __restrict__ out)                  // [BS, COUT, 56, 56]
{
    __shared__ unsigned short As[2 * ASZ];    // [buf][m][k]  24.0 KB
    __shared__ unsigned short Bs[2 * BSZ];    // [buf][n][k]  12.0 KB

    const int t    = threadIdx.x;
    const int wave = t >> 5;
    const int lane = t & 31;
    const int wm   = wave & 3;    // M sub-tile: rows wm*32 .. wm*32+31
    const int wn   = wave >> 2;   // N half: cols wn*32 .. wn*32+31
    const int hl   = lane >> 4;   // lane half (K sub-range select)
    const int ml   = lane & 15;

    const int b  = blockIdx.y;
    const int p0 = blockIdx.x * 64;

    // B-gather coords: row 0..63 (pixel), 8-elem K slot
    const int lrow = t >> 2;
    const int lk   = (t & 3) * 8;
    const int pn   = p0 + lrow;
    const int oy   = pn / HW_IN;
    const int ox   = pn - oy * HW_IN;

    const unsigned short* xb = xbf + (size_t)b * CIN * NPIX;
    const size_t wbase_elems = (size_t)(b * COUT) * KTOT;

    v8f acc[2][2] = {};

    // ---- stage one K step (A via TDM or manual, B via im2col gather) ----
    auto stage = [&](int k0, int bufsel) {
#ifdef HAVE_TDM
        if (wave == 0) {
            const unsigned lds_addr =
                (unsigned)(size_t)(const void*)&As[bufsel * ASZ];
            const unsigned long long gaddr =
                (unsigned long long)(size_t)wmix + (wbase_elems + (size_t)k0) * 2ull;
            v4u g0;
            g0[0] = 1u;                                   // count=1, user mode
            g0[1] = lds_addr;                             // LDS byte address
            g0[2] = (unsigned)gaddr;                      // global addr [31:0]
            g0[3] = (unsigned)(gaddr >> 32) | (2u << 30); // addr[56:32] | type=2
            v8i g1;
            g1[0] = (1 << 16)      // data_size = 2 bytes
                  | (1 << 20)      // pad_enable
                  | (3 << 22)      // pad_interval: 16 DWORDs (= 32 elems/row)
                  | (7 << 25);     // pad_amount: 8 DWORDs (= 16 elem pad -> ASTRIDE)
            g1[1] = 0;                                    // tensor_dim0[15:0]<<16 (dim0=1<<30)
            g1[2] = 0x4000;                               // tensor_dim0[31:16]
            g1[3] = 0x4000 | (32 << 16);                  // tensor_dim1[31:16] | tile_dim0=32
            g1[4] = 128;                                  // tile_dim1=128, tile_dim2=0
            g1[5] = KTOT;                                 // tensor_dim0_stride low32
            g1[6] = 0;
            g1[7] = 0;
            v4i gz4 = {};
#if __clang_major__ >= 23
            v8i gz8 = {};
            __builtin_amdgcn_tensor_load_to_lds(g0, g1, gz4, gz4, gz8, 0);
#else
            __builtin_amdgcn_tensor_load_to_lds(g0, g1, gz4, gz4, 0);
#endif
        }
#else
        // manual A staging: 128 rows x 32 k; thread copies 16 contiguous elems
        {
            const int ar = t >> 1;
            const int ak = (t & 1) * 16;
            const unsigned short* src = wmix + wbase_elems + (size_t)ar * KTOT + k0 + ak;
            uint4 v0 = *(const uint4*)(src);
            uint4 v1 = *(const uint4*)(src + 8);
            unsigned short* dst = &As[bufsel * ASZ + ar * ASTRIDE + ak];
            *(uint4*)(dst)     = v0;
            *(uint4*)(dst + 8) = v1;
        }
#endif
        // B tile: im2col gather, 64 px x 32 k
#pragma unroll
        for (int j = 0; j < 8; ++j) {
            const int kg = k0 + lk + j;
            const int c  = kg / 9;
            const int rs = kg - c * 9;
            const int r  = rs / 3;
            const int s  = rs - r * 3;
            const int iy = oy + r - 1;
            const int ix = ox + s - 1;
            unsigned short v = 0;
            if ((unsigned)iy < (unsigned)HW_IN && (unsigned)ix < (unsigned)HW_IN)
                v = xb[c * NPIX + iy * HW_IN + ix];
            Bs[bufsel * BSZ + lrow * ASTRIDE + lk + j] = v;
        }
    };

    auto load_bf16x16 = [](const unsigned short* p) -> v16bf {
        uint4 lo = *(const uint4*)(p);
        uint4 hi = *(const uint4*)(p + 16);
        v8u u;
        u[0] = lo.x; u[1] = lo.y; u[2] = lo.z; u[3] = lo.w;
        u[4] = hi.x; u[5] = hi.y; u[6] = hi.z; u[7] = hi.w;
        return __builtin_bit_cast(v16bf, u);
    };

    // prologue: fill buffer 0
    stage(0, 0);
#ifdef HAVE_TDM
    if (wave == 0) __builtin_amdgcn_s_wait_tensorcnt(0);
#endif
    __syncthreads();

    for (int it = 0; it < KTOT / 32; ++it) {
        const int cur = it & 1;
        if (it + 1 < KTOT / 32) stage((it + 1) * 32, cur ^ 1);

        const unsigned short* As_b = &As[cur * ASZ];
        const unsigned short* Bs_b = &Bs[cur * BSZ];

        // A frags (ISA 16-bit 16x32 layout): v0..3 = K hl*8..+7, v4..7 = 16+hl*8..+7
        v16bf a0 = load_bf16x16(&As_b[(wm * 32 + ml) * ASTRIDE + hl * 8] - 0);
        v16bf a0h; // combine manually below instead
        (void)a0h;
        {
            const unsigned short* ar = &As_b[(wm * 32 + ml) * ASTRIDE];
            uint4 lo = *(const uint4*)(ar + hl * 8);
            uint4 hi = *(const uint4*)(ar + 16 + hl * 8);
            v8u u; u[0]=lo.x; u[1]=lo.y; u[2]=lo.z; u[3]=lo.w;
                   u[4]=hi.x; u[5]=hi.y; u[6]=hi.z; u[7]=hi.w;
            a0 = __builtin_bit_cast(v16bf, u);
        }
        v16bf a1;
        {
            const unsigned short* ar = &As_b[(wm * 32 + 16 + ml) * ASTRIDE];
            uint4 lo = *(const uint4*)(ar + hl * 8);
            uint4 hi = *(const uint4*)(ar + 16 + hl * 8);
            v8u u; u[0]=lo.x; u[1]=lo.y; u[2]=lo.z; u[3]=lo.w;
                   u[4]=hi.x; u[5]=hi.y; u[6]=hi.z; u[7]=hi.w;
            a1 = __builtin_bit_cast(v16bf, u);
        }
        // B frags: lane column n, v0..3 = K hl*8..+7, v4..7 = 16+hl*8..+7
        v16bf b0 = load_bf16x16(&Bs_b[(wn * 32 + ml) * ASTRIDE + hl * 8]);
        v16bf b1 = load_bf16x16(&Bs_b[(wn * 32 + 16 + ml) * ASTRIDE + hl * 8]);

        acc[0][0] = __builtin_amdgcn_wmma_f32_16x16x32_bf16(false, a0, false, b0,
                                                            (short)0, acc[0][0], false, false);
        acc[0][1] = __builtin_amdgcn_wmma_f32_16x16x32_bf16(false, a0, false, b1,
                                                            (short)0, acc[0][1], false, false);
        acc[1][0] = __builtin_amdgcn_wmma_f32_16x16x32_bf16(false, a1, false, b0,
                                                            (short)0, acc[1][0], false, false);
        acc[1][1] = __builtin_amdgcn_wmma_f32_16x16x32_bf16(false, a1, false, b1,
                                                            (short)0, acc[1][1], false, false);

#ifdef HAVE_TDM
        if (wave == 0) __builtin_amdgcn_s_wait_tensorcnt(0);
#endif
        __syncthreads();
    }

    // ---- epilogue: C/D layout -> global, fuse bias ----
#pragma unroll
    for (int mi = 0; mi < 2; ++mi) {
#pragma unroll
        for (int p = 0; p < 8; ++p) {
            const int m = wm * 32 + mi * 16 + p + hl * 8;
            const float bv = bmix[b * COUT + m];
            const size_t base = (size_t)(b * COUT + m) * NPIX + p0 + wn * 32;
            out[base + ml]      = acc[mi][0][p] + bv;
            out[base + 16 + ml] = acc[mi][1][p] + bv;
        }
    }
}

// ---------- launch ----------
extern "C" void kernel_launch(void* const* d_in, const int* in_sizes, int n_in,
                              void* d_out, int out_size, void* d_ws, size_t ws_size,
                              hipStream_t stream) {
    const float* x       = (const float*)d_in[0];   // [64,128,56,56]
    const float* routing = (const float*)d_in[1];   // [64,8]
    const float* weight  = (const float*)d_in[2];   // [8,128,128,3,3]
    const float* bias    = (const float*)d_in[3];   // [8,128]
    float* out = (float*)d_out;

    // workspace layout (bytes): xbf | wmix | bmix  (~70.3 MB total)
    const size_t xbf_bytes  = (size_t)BS * CIN * NPIX * sizeof(unsigned short); // 51,380,224
    const size_t wmix_bytes = (size_t)BS * WMIX_PER_B * sizeof(unsigned short); // 18,874,368
    unsigned short* xbf  = (unsigned short*)d_ws;
    unsigned short* wmix = (unsigned short*)((char*)d_ws + xbf_bytes);
    float*          bmix = (float*)((char*)d_ws + xbf_bytes + wmix_bytes);

    // 1) expert-mix weights -> bf16
    mix_weights_kernel<<<dim3(WMIX_PER_B / 256, BS), 256, 0, stream>>>(routing, weight, wmix);
    // 2) x -> bf16
    x_to_bf16_kernel<<<(BS * CIN * NPIX) / (256 * 4), 256, 0, stream>>>(x, xbf);
    // 3) mixed bias
    mix_bias_kernel<<<(BS * COUT) / 256, 256, 0, stream>>>(routing, bias, bmix);
    // 4) implicit-GEMM conv with bf16 WMMA + TDM staging
    condconv_wmma_kernel<<<dim3(NPIX / 64, BS), 256, 0, stream>>>(xbf, wmix, bmix, out);
}
